// RNN_63591285785311
// MI455X (gfx1250) — matrix operand order repaired and checked by reading, not compiled, not run on hardware
//
#include <hip/hip_runtime.h>

typedef unsigned short u16;
typedef __attribute__((ext_vector_type(16))) __bf16 v16bf;
typedef __attribute__((ext_vector_type(8)))  float  v8f;

union Frag { uint4 q[2]; v16bf v; };

#define HD 1024
#define BB 64
#define SS 512

__device__ __forceinline__ unsigned bfbits(float f) {
    unsigned u = __float_as_uint(f);
    return (u + 0x7FFFu + ((u >> 16) & 1u)) >> 16;   // RNE
}
__device__ __forceinline__ unsigned pkbf(float a, float b) {
#if __has_builtin(__builtin_amdgcn_cvt_pk_bf16_f32)
    typedef __attribute__((ext_vector_type(2))) __bf16 v2bf;
    v2bf r = __builtin_amdgcn_cvt_pk_bf16_f32(a, b);
    return __builtin_bit_cast(unsigned, r);
#else
    return (bfbits(a) & 0xFFFFu) | ((bfbits(b) & 0xFFFFu) << 16);
#endif
}
__device__ __forceinline__ float bf2f(u16 h) {
    return __uint_as_float(((unsigned)h) << 16);
}
__device__ __forceinline__ v8f wmma_bf16(v16bf a, v16bf b, v8f c) {
    return __builtin_amdgcn_wmma_f32_16x16x32_bf16(false, a, false, b, (short)0, c, false, false);
}

// ---------------------------------------------------------------------------
// Kernel 1: convert weights + emb f32 -> bf16 (packed), reset barrier counter
// ---------------------------------------------------------------------------
__global__ void __launch_bounds__(256) rnn_prep(
    const float2* __restrict__ Vw, const float2* __restrict__ Ww,
    const float2* __restrict__ Uw, const float2* __restrict__ emb,
    unsigned* __restrict__ Vwb, unsigned* __restrict__ Wwb,
    unsigned* __restrict__ Uwb, unsigned* __restrict__ embb,
    unsigned* __restrict__ cnt)
{
    const int tid = blockIdx.x * 256 + threadIdx.x;
    if (tid == 0) *cnt = 0u;
    const int NW = HD * HD / 2;
    for (int i = tid; i < NW; i += gridDim.x * 256) {
        float2 v = Vw[i]; Vwb[i] = pkbf(v.x, v.y);
        float2 w = Ww[i]; Wwb[i] = pkbf(w.x, w.y);
        float2 u = Uw[i]; Uwb[i] = pkbf(u.x, u.y);
    }
    const int NE = BB * SS * HD / 2;
    for (int i = tid; i < NE; i += gridDim.x * 256) {
        float2 e = emb[i]; embb[i] = pkbf(e.x, e.y);
    }
}

// ---------------------------------------------------------------------------
// Kernel 2: P[t][b][n] = bf16( emb @ Vw^T + (Vb + Wb) )     (m = b*S + t)
// grid 2048 x 256; each wave computes a 32(M) x 64(N) tile
// ---------------------------------------------------------------------------
__global__ void __launch_bounds__(256) rnn_gemm_pre(
    const u16* __restrict__ X,     // embbf as [B*S][H]
    const u16* __restrict__ Wt,    // Vw bf16 row-major [n][k]
    const float* __restrict__ b1, const float* __restrict__ b2,
    u16* __restrict__ P)           // [S][B][H] bf16
{
    const int lane = threadIdx.x & 31, wid = threadIdx.x >> 5;
    const int waveId = blockIdx.x * 8 + wid;         // 0..16383
    const int m0 = (waveId >> 4) * 32;               // 1024 M-tiles
    const int n0 = (waveId & 15) * 64;               // 16 N-tiles
    const int la = lane & 15, lh = lane >> 4;

    v8f acc[8] = {};
    const u16* A0 = X + (size_t)(m0 + la) * HD;
    const u16* A1 = X + (size_t)(m0 + 16 + la) * HD;
    for (int kt = 0; kt < HD; kt += 32) {
        Frag a0, a1, b[4];
        const int ka = kt + lh * 8;
        a0.q[0] = *(const uint4*)(A0 + ka);
        a0.q[1] = *(const uint4*)(A0 + ka + 16);
        a1.q[0] = *(const uint4*)(A1 + ka);
        a1.q[1] = *(const uint4*)(A1 + ka + 16);
        const int kb = kt + lh * 8;
#pragma unroll
        for (int c = 0; c < 4; ++c) {
            const u16* Br = Wt + (size_t)(n0 + c * 16 + la) * HD + kb;
            b[c].q[0] = *(const uint4*)(Br);
            b[c].q[1] = *(const uint4*)(Br + 16);
        }
#pragma unroll
        for (int c = 0; c < 4; ++c) acc[c]     = wmma_bf16(a0.v, b[c].v, acc[c]);
#pragma unroll
        for (int c = 0; c < 4; ++c) acc[4 + c] = wmma_bf16(a1.v, b[c].v, acc[4 + c]);
    }
#pragma unroll
    for (int c = 0; c < 4; ++c) {
        const int n = n0 + c * 16 + la;
        const float bs = b1[n] + b2[n];
#pragma unroll
        for (int j = 0; j < 8; ++j) {
            int m = m0 + j + lh * 8;                 // m = b*S + t
            int bi = m >> 9, ti = m & 511;
            P[((size_t)ti * BB + bi) * HD + n] = (u16)bfbits(acc[c][j] + bs);
            m += 16;
            bi = m >> 9; ti = m & 511;
            P[((size_t)ti * BB + bi) * HD + n] = (u16)bfbits(acc[4 + c][j] + bs);
        }
    }
}

// ---------------------------------------------------------------------------
// Kernel 3: s_0 = relu(P_0)  — exact on bf16 via sign bit
// ---------------------------------------------------------------------------
__global__ void __launch_bounds__(256) rnn_init(
    const u16* __restrict__ Pb, u16* __restrict__ Sb)
{
    const int i = blockIdx.x * 256 + threadIdx.x;    // B*H = 65536 total
    const u16 h = Pb[i];
    Sb[i] = (h & 0x8000u) ? (u16)0 : h;
}

// ---------------------------------------------------------------------------
// Kernel 4: persistent recurrence.  32 blocks; block owns 32 N-columns,
// stages its 32 rows of Ww (64KB bf16) in LDS once; 511 steps with a
// grid-wide release/acquire barrier between steps.  4 independent wmma
// accumulator chains over K for ILP.
// ---------------------------------------------------------------------------
__global__ void __launch_bounds__(256, 1) rnn_scan(
    const u16* __restrict__ Wwb, const u16* __restrict__ Pb,
    u16* __restrict__ Sb, unsigned* __restrict__ cnt)
{
    __shared__ u16 smem[32 * HD];                    // 64KB: Ww rows [32*blk, +32)
    const int lane = threadIdx.x & 31, wid = threadIdx.x >> 5;
    const int la = lane & 15, lh = lane >> 4;
    const int m0   = (wid & 3) * 16;                 // 4 M-tiles cover B=64
    const int nloc = (wid >> 2) * 16;                // 0 or 16
    const int n    = blockIdx.x * 32 + nloc + la;    // global output column

    {   // stage weight slice into LDS
        const uint4* src = (const uint4*)(Wwb + (size_t)blockIdx.x * 32 * HD);
        uint4* dst = (uint4*)smem;
#pragma unroll
        for (int i = 0; i < 16; ++i)
            dst[threadIdx.x + i * 256] = src[threadIdx.x + i * 256];
    }
    __syncthreads();

    const u16* Bls = smem + (size_t)(nloc + la) * HD;
    for (int t = 1; t < SS; ++t) {
        v8f acc[4] = {};
        const u16* Ar = Sb + ((size_t)(t - 1) * BB + m0 + la) * HD;
#pragma unroll 2
        for (int kt = 0; kt < HD; kt += 128) {
#pragma unroll
            for (int i = 0; i < 4; ++i) {            // 4 independent chains
                const int k = kt + i * 32;
                Frag a, b;
                a.q[0] = *(const uint4*)(Ar + k + lh * 8);
                a.q[1] = *(const uint4*)(Ar + k + lh * 8 + 16);
                b.q[0] = *(const uint4*)(Bls + k + lh * 8);        // ds_load_b128
                b.q[1] = *(const uint4*)(Bls + k + lh * 8 + 16);   // ds_load_b128
                acc[i] = wmma_bf16(a.v, b.v, acc[i]);
            }
        }
        const v8f r = (acc[0] + acc[1]) + (acc[2] + acc[3]);
        const u16* Pt = Pb + (size_t)t * BB * HD;
        u16*       St = Sb + (size_t)t * BB * HD;
#pragma unroll
        for (int j = 0; j < 8; ++j) {
            const int m = m0 + j + lh * 8;
            const float v = fmaxf(r[j] + bf2f(Pt[(size_t)m * HD + n]), 0.f);
            St[(size_t)m * HD + n] = (u16)bfbits(v);
        }
        // grid-wide barrier: release writes of s_t, acquire before reading it
        __syncthreads();
        if (threadIdx.x == 0) {
            __threadfence();
            __hip_atomic_fetch_add(cnt, 1u, __ATOMIC_RELEASE, __HIP_MEMORY_SCOPE_AGENT);
            const unsigned tgt = 32u * (unsigned)t;
            while (__hip_atomic_load(cnt, __ATOMIC_ACQUIRE, __HIP_MEMORY_SCOPE_AGENT) < tgt)
                __builtin_amdgcn_s_sleep(1);
        }
        __syncthreads();
    }
}

// ---------------------------------------------------------------------------
// Kernel 5: out[b][t][n] = S @ Uw^T + Ub     (m = t*B + b), f32 output
// ---------------------------------------------------------------------------
__global__ void __launch_bounds__(256) rnn_gemm_out(
    const u16* __restrict__ Sb,   // [S][B][H] bf16
    const u16* __restrict__ Ut,   // Uw bf16 row-major [n][k]
    const float* __restrict__ Ub, float* __restrict__ out)
{
    const int lane = threadIdx.x & 31, wid = threadIdx.x >> 5;
    const int waveId = blockIdx.x * 8 + wid;
    const int m0 = (waveId >> 4) * 32;
    const int n0 = (waveId & 15) * 64;
    const int la = lane & 15, lh = lane >> 4;

    v8f acc[8] = {};
    const u16* A0 = Sb + (size_t)(m0 + la) * HD;
    const u16* A1 = Sb + (size_t)(m0 + 16 + la) * HD;
    for (int kt = 0; kt < HD; kt += 32) {
        Frag a0, a1, b[4];
        const int ka = kt + lh * 8;
        a0.q[0] = *(const uint4*)(A0 + ka);
        a0.q[1] = *(const uint4*)(A0 + ka + 16);
        a1.q[0] = *(const uint4*)(A1 + ka);
        a1.q[1] = *(const uint4*)(A1 + ka + 16);
        const int kb = kt + lh * 8;
#pragma unroll
        for (int c = 0; c < 4; ++c) {
            const u16* Br = Ut + (size_t)(n0 + c * 16 + la) * HD + kb;
            b[c].q[0] = *(const uint4*)(Br);
            b[c].q[1] = *(const uint4*)(Br + 16);
        }
#pragma unroll
        for (int c = 0; c < 4; ++c) acc[c]     = wmma_bf16(a0.v, b[c].v, acc[c]);
#pragma unroll
        for (int c = 0; c < 4; ++c) acc[4 + c] = wmma_bf16(a1.v, b[c].v, acc[4 + c]);
    }
#pragma unroll
    for (int c = 0; c < 4; ++c) {
        const int n = n0 + c * 16 + la;
        const float bs = Ub[n];
#pragma unroll
        for (int j = 0; j < 8; ++j) {
            int m = m0 + j + lh * 8;                 // m = t*B + b
            int bi = m & 63, ti = m >> 6;
            out[((size_t)bi * SS + ti) * HD + n] = acc[c][j] + bs;
            m += 16;
            bi = m & 63; ti = m >> 6;
            out[((size_t)bi * SS + ti) * HD + n] = acc[4 + c][j] + bs;
        }
    }
}

// ---------------------------------------------------------------------------
extern "C" void kernel_launch(void* const* d_in, const int* in_sizes, int n_in,
                              void* d_out, int out_size, void* d_ws, size_t ws_size,
                              hipStream_t stream) {
    const float* emb = (const float*)d_in[0];
    const float* Vw  = (const float*)d_in[1];
    const float* Vb  = (const float*)d_in[2];
    const float* Ww  = (const float*)d_in[3];
    const float* Wb  = (const float*)d_in[4];
    const float* Uw  = (const float*)d_in[5];
    const float* Ub  = (const float*)d_in[6];
    float* out = (float*)d_out;

    char* ws = (char*)d_ws;
    u16*      Wwbf  = (u16*)(ws);                                 // 2MB
    u16*      Uwbf  = (u16*)(ws + ((size_t)2 << 20));             // 2MB
    u16*      Vwbf  = (u16*)(ws + ((size_t)4 << 20));             // 2MB
    unsigned* cnt   = (unsigned*)(ws + ((size_t)6 << 20));
    u16*      embbf = (u16*)(ws + ((size_t)8 << 20));             // 64MB  [B*S][H] bf16
    u16*      Pb    = (u16*)(ws + ((size_t)72 << 20));            // 64MB  [S][B][H] bf16
    u16*      Sb    = (u16*)(ws + ((size_t)136 << 20));           // 64MB  [S][B][H] bf16

    rnn_prep<<<8192, 256, 0, stream>>>(
        (const float2*)Vw, (const float2*)Ww, (const float2*)Uw, (const float2*)emb,
        (unsigned*)Vwbf, (unsigned*)Wwbf, (unsigned*)Uwbf, (unsigned*)embbf, cnt);
    rnn_gemm_pre<<<2048, 256, 0, stream>>>(embbf, Vwbf, Vb, Wb, Pb);
    rnn_init<<<(BB * HD) / 256, 256, 0, stream>>>(Pb, Sb);
    rnn_scan<<<32, 256, 0, stream>>>(Wwbf, Pb, Sb, cnt);
    rnn_gemm_out<<<2048, 256, 0, stream>>>(Sb, Uwbf, Ub, out);
}
